// SelfAttention_81131932221963
// MI455X (gfx1250) — compile-verified, matrix operand order checked by reading, not run on hardware
//
#include <hip/hip_runtime.h>
#include <hip/hip_bf16.h>

// ---------------------------------------------------------------------------
// SelfAttention for MI455X (gfx1250): bf16 WMMA pipeline.
//   x:(8,512,4096) wq/wk:(64,512) wv:(512,512)  ->  out:(8,512,4096), att:(8,4096,4096)
//
// Pipeline:
//   k_xt       : x (B,C,S) f32 -> xt (B,S,C) bf16            (LDS tiled transpose)
//   k_cvt      : weights f32 -> bf16
//   k_proj_qk  : qt/kt (B,S,64) bf16      [WMMA]
//   k_proj_v   : vbf (B,C,S)  bf16        [WMMA]
//   k_energy   : att_f32 = Qt x Kt^T      [WMMA] -> d_out attention region
//   k_softmax  : normalize rows in place, also emit bf16 copy att_bf -> ws
//   k_out      : out = gamma*(V x Att^T) + x   [WMMA, pure bf16 operands]
//
// Workspace requirement: ~344 MB (bf16 att copy is 268 MB).
// ---------------------------------------------------------------------------

typedef __attribute__((ext_vector_type(16))) __bf16 v16bf;
typedef __attribute__((ext_vector_type(8)))  float  v8f;

#define B_   8
#define C_   512
#define CQ_  64
#define S_   4096

__device__ __forceinline__ unsigned short f2bf(float f) {
    __bf16 h = (__bf16)f;                       // RNE convert
    return __builtin_bit_cast(unsigned short, h);
}

// ---------------------------------------------------------------------------
// 0a. transpose + convert: xt[b,s,c] = bf16(x[b,c,s])
// ---------------------------------------------------------------------------
__global__ void k_xt(const float* __restrict__ x, unsigned short* __restrict__ xt) {
    __shared__ float tile[16][17];
    const int b  = blockIdx.z;
    const int s0 = blockIdx.x * 16;
    const int c0 = blockIdx.y * 16;
    const int tx = threadIdx.x, ty = threadIdx.y;
    tile[ty][tx] = x[((size_t)b * C_ + c0 + ty) * S_ + s0 + tx];
    __syncthreads();
    xt[((size_t)b * S_ + s0 + ty) * C_ + c0 + tx] = f2bf(tile[tx][ty]);
}

// ---------------------------------------------------------------------------
// 0b. generic f32 -> bf16 convert (weight matrices)
// ---------------------------------------------------------------------------
__global__ void k_cvt(const float* __restrict__ src, unsigned short* __restrict__ dst, int n) {
    int i = blockIdx.x * blockDim.x + threadIdx.x;
    if (i < n) dst[i] = f2bf(src[i]);
}

// ---------------------------------------------------------------------------
// 1. Q/K projection:  out[b,s,o] = bf16( sum_c xt[b,s,c]*w[o,c] + bias[o] )
//    One wave: 16 s-rows x 64 o-cols (4 accumulators), K = 512.
// ---------------------------------------------------------------------------
__global__ void k_proj_qk(const unsigned short* __restrict__ xt,
                          const unsigned short* __restrict__ w,
                          const float* __restrict__ bias,
                          unsigned short* __restrict__ outp) {
    const int wave  = blockIdx.x * (blockDim.x >> 5) + (threadIdx.x >> 5);
    const int lane  = threadIdx.x & 31;
    const int b     = wave / (S_ / 16);
    const int s0    = (wave % (S_ / 16)) * 16;
    const int l15   = lane & 15;
    const int khalf = (lane >> 4) << 4;

    v8f acc[4];
#pragma unroll
    for (int j = 0; j < 4; ++j) acc[j] = (v8f){0.f,0.f,0.f,0.f,0.f,0.f,0.f,0.f};

    const unsigned short* arow = xt + ((size_t)b * S_ + s0 + l15) * C_;
    for (int kc = 0; kc < C_; kc += 32) {
        v16bf a = *(const v16bf*)(arow + kc + khalf);
#pragma unroll
        for (int j = 0; j < 4; ++j) {
            v16bf bm = *(const v16bf*)(w + (size_t)(j * 16 + l15) * C_ + kc + khalf);
            acc[j] = __builtin_amdgcn_wmma_f32_16x16x32_bf16(
                false, a, false, bm, (short)0, acc[j], false, false);
        }
    }

    const int n  = l15;
    const int mh = (lane >> 4) * 8;
#pragma unroll
    for (int j = 0; j < 4; ++j) {
        const int   o  = j * 16 + n;
        const float bv = bias[o];
#pragma unroll
        for (int g = 0; g < 8; ++g) {
            const int m = mh + g;
            outp[((size_t)b * S_ + s0 + m) * CQ_ + o] = f2bf(acc[j][g] + bv);
        }
    }
}

// ---------------------------------------------------------------------------
// 2. V projection: vbf[b,o,s] = bf16( sum_c wv[o,c]*xt[b,s,c] + bv[o] )
//    One wave: 16 o-rows x 64 s-cols, K = 512.
// ---------------------------------------------------------------------------
__global__ void k_proj_v(const unsigned short* __restrict__ xt,
                         const unsigned short* __restrict__ wv,
                         const float* __restrict__ bv,
                         unsigned short* __restrict__ vbf) {
    const int wave  = blockIdx.x * (blockDim.x >> 5) + (threadIdx.x >> 5);
    const int lane  = threadIdx.x & 31;
    const int tpb   = (C_ / 16) * (S_ / 64);         // 2048
    const int b     = wave / tpb;
    const int r     = wave % tpb;
    const int o0    = (r / (S_ / 64)) * 16;
    const int s0    = (r % (S_ / 64)) * 64;
    const int l15   = lane & 15;
    const int khalf = (lane >> 4) << 4;

    v8f acc[4];
#pragma unroll
    for (int j = 0; j < 4; ++j) acc[j] = (v8f){0.f,0.f,0.f,0.f,0.f,0.f,0.f,0.f};

    const unsigned short* arow = wv + (size_t)(o0 + l15) * C_;
    for (int kc = 0; kc < C_; kc += 32) {
        v16bf a = *(const v16bf*)(arow + kc + khalf);
#pragma unroll
        for (int j = 0; j < 4; ++j) {
            v16bf bm = *(const v16bf*)(xt + ((size_t)b * S_ + s0 + j * 16 + l15) * C_ + kc + khalf);
            acc[j] = __builtin_amdgcn_wmma_f32_16x16x32_bf16(
                false, a, false, bm, (short)0, acc[j], false, false);
        }
    }

    const int n  = l15;
    const int mh = (lane >> 4) * 8;
#pragma unroll
    for (int j = 0; j < 4; ++j) {
#pragma unroll
        for (int g = 0; g < 8; ++g) {
            const int m = mh + g;
            vbf[((size_t)b * C_ + o0 + m) * S_ + s0 + j * 16 + n] = f2bf(acc[j][g] + bv[o0 + m]);
        }
    }
}

// ---------------------------------------------------------------------------
// 3. Energy: att[b,s,t] = sum_c qt[b,s,c]*kt[b,t,c]   (fp32, pre-softmax)
//    One wave: 16 s-rows x 64 t-cols, K = 64.
// ---------------------------------------------------------------------------
__global__ void k_energy(const unsigned short* __restrict__ qt,
                         const unsigned short* __restrict__ kt,
                         float* __restrict__ att) {
    const int wave  = blockIdx.x * (blockDim.x >> 5) + (threadIdx.x >> 5);
    const int lane  = threadIdx.x & 31;
    const int tpb   = (S_ / 16) * (S_ / 64);         // 16384
    const int b     = wave / tpb;
    const int r     = wave % tpb;
    const int s0    = (r / (S_ / 64)) * 16;
    const int t0    = (r % (S_ / 64)) * 64;
    const int l15   = lane & 15;
    const int khalf = (lane >> 4) << 4;

    v8f acc[4];
#pragma unroll
    for (int j = 0; j < 4; ++j) acc[j] = (v8f){0.f,0.f,0.f,0.f,0.f,0.f,0.f,0.f};

    const unsigned short* arow = qt + ((size_t)b * S_ + s0 + l15) * CQ_;
#pragma unroll
    for (int kc = 0; kc < CQ_; kc += 32) {
        v16bf a = *(const v16bf*)(arow + kc + khalf);
#pragma unroll
        for (int j = 0; j < 4; ++j) {
            v16bf bm = *(const v16bf*)(kt + ((size_t)b * S_ + t0 + j * 16 + l15) * CQ_ + kc + khalf);
            acc[j] = __builtin_amdgcn_wmma_f32_16x16x32_bf16(
                false, a, false, bm, (short)0, acc[j], false, false);
        }
    }

    const int n  = l15;
    const int mh = (lane >> 4) * 8;
#pragma unroll
    for (int j = 0; j < 4; ++j) {
#pragma unroll
        for (int g = 0; g < 8; ++g) {
            const int m = mh + g;
            att[((size_t)b * S_ + s0 + m) * S_ + t0 + j * 16 + n] = acc[j][g];
        }
    }
}

// ---------------------------------------------------------------------------
// 4. Row softmax over the last axis, in place (fp32) + bf16 copy for k_out.
//    One 256-thread block per row.
// ---------------------------------------------------------------------------
__global__ void k_softmax(float* __restrict__ att, unsigned short* __restrict__ attbf) {
    __shared__ float red[256];
    const int row = blockIdx.x;                      // b*S + t_row
    float*          p  = att   + (size_t)row * S_;
    unsigned short* pb = attbf + (size_t)row * S_;
    const int t = threadIdx.x;

    float vals[S_ / 256];
    float mx = -3.402823466e38f;
#pragma unroll
    for (int i = 0; i < S_ / 256; ++i) {
        vals[i] = p[t + i * 256];
        mx = fmaxf(mx, vals[i]);
    }
    red[t] = mx; __syncthreads();
    for (int off = 128; off > 0; off >>= 1) {
        if (t < off) red[t] = fmaxf(red[t], red[t + off]);
        __syncthreads();
    }
    mx = red[0]; __syncthreads();

    float sum = 0.f;
#pragma unroll
    for (int i = 0; i < S_ / 256; ++i) {
        vals[i] = __expf(vals[i] - mx);
        sum += vals[i];
    }
    red[t] = sum; __syncthreads();
    for (int off = 128; off > 0; off >>= 1) {
        if (t < off) red[t] += red[t + off];
        __syncthreads();
    }
    const float inv = 1.0f / red[0];
#pragma unroll
    for (int i = 0; i < S_ / 256; ++i) {
        const float a = vals[i] * inv;
        p[t + i * 256]  = a;
#if __has_builtin(__builtin_amdgcn_cvt_pk_bf16_f32)
        // packed variant exists but lanes here write strided singles; scalar is fine
        pb[t + i * 256] = f2bf(a);
#else
        pb[t + i * 256] = f2bf(a);
#endif
    }
}

// ---------------------------------------------------------------------------
// 5. out[b,c,t] = gamma * sum_s vbf[b,c,s]*attbf[b,t,s] + x[b,c,t]
//    One wave: 16 c-rows x 64 t-cols, K = 4096, pure bf16 operands.
//    t-group-major tile order: concurrent waves/blocks share the attention
//    rows (B operand) so attention streams from HBM ~once via L2 reuse.
// ---------------------------------------------------------------------------
__global__ void k_out(const unsigned short* __restrict__ vbf,
                      const unsigned short* __restrict__ attbf,
                      const float* __restrict__ x,
                      const float* __restrict__ gamma,
                      float* __restrict__ out) {
    const int wave  = blockIdx.x * (blockDim.x >> 5) + (threadIdx.x >> 5);
    const int lane  = threadIdx.x & 31;
    const int tpb   = (C_ / 16) * (S_ / 64);         // 2048
    const int b     = wave / tpb;
    const int r     = wave % tpb;
    const int t0    = (r / (C_ / 16)) * 64;          // t-group major
    const int c0    = (r % (C_ / 16)) * 16;
    const int l15   = lane & 15;
    const int khalf = (lane >> 4) << 4;

    v8f acc[4];
#pragma unroll
    for (int j = 0; j < 4; ++j) acc[j] = (v8f){0.f,0.f,0.f,0.f,0.f,0.f,0.f,0.f};

    const unsigned short* arow = vbf + ((size_t)b * C_ + c0 + l15) * S_;
    const unsigned short* brow[4];
#pragma unroll
    for (int j = 0; j < 4; ++j)
        brow[j] = attbf + ((size_t)b * S_ + t0 + j * 16 + l15) * S_;

    for (int kc = 0; kc < S_; kc += 32) {
        v16bf a = *(const v16bf*)(arow + kc + khalf);
#pragma unroll
        for (int j = 0; j < 4; ++j) {
            v16bf bm = *(const v16bf*)(brow[j] + kc + khalf);
            acc[j] = __builtin_amdgcn_wmma_f32_16x16x32_bf16(
                false, a, false, bm, (short)0, acc[j], false, false);
        }
    }

    const float g_ = gamma[0];
    const int n  = l15;
    const int mh = (lane >> 4) * 8;
#pragma unroll
    for (int j = 0; j < 4; ++j) {
#pragma unroll
        for (int gi = 0; gi < 8; ++gi) {
            const int    m   = mh + gi;
            const size_t idx = ((size_t)b * C_ + c0 + m) * S_ + t0 + j * 16 + n;
            out[idx] = g_ * acc[j][gi] + x[idx];
        }
    }
}

// ---------------------------------------------------------------------------
// Host launcher
// ---------------------------------------------------------------------------
extern "C" void kernel_launch(void* const* d_in, const int* in_sizes, int n_in,
                              void* d_out, int out_size, void* d_ws, size_t ws_size,
                              hipStream_t stream) {
    const float* x     = (const float*)d_in[0];
    const float* wq    = (const float*)d_in[1];
    const float* bq    = (const float*)d_in[2];
    const float* wk    = (const float*)d_in[3];
    const float* bk    = (const float*)d_in[4];
    const float* wv    = (const float*)d_in[5];
    const float* bv    = (const float*)d_in[6];
    const float* gamma = (const float*)d_in[7];

    float* out = (float*)d_out;
    float* att = out + (size_t)B_ * C_ * S_;         // (B,S,S) attention output

    // Workspace carve-up (bf16 buffers as ushort); total ~344 MB
    char* ws = (char*)d_ws;
    size_t cur = 0;
    unsigned short* xt    = (unsigned short*)(ws + cur); cur += (size_t)B_ * S_ * C_ * 2;   // 32 MB
    unsigned short* wqb   = (unsigned short*)(ws + cur); cur += (size_t)CQ_ * C_ * 2;       // 64 KB
    unsigned short* wkb   = (unsigned short*)(ws + cur); cur += (size_t)CQ_ * C_ * 2;       // 64 KB
    unsigned short* wvb   = (unsigned short*)(ws + cur); cur += (size_t)C_ * C_ * 2;        // 512 KB
    unsigned short* qt    = (unsigned short*)(ws + cur); cur += (size_t)B_ * S_ * CQ_ * 2;  // 4 MB
    unsigned short* kt    = (unsigned short*)(ws + cur); cur += (size_t)B_ * S_ * CQ_ * 2;  // 4 MB
    unsigned short* vbf   = (unsigned short*)(ws + cur); cur += (size_t)B_ * C_ * S_ * 2;   // 32 MB
    unsigned short* attbf = (unsigned short*)(ws + cur); cur += (size_t)B_ * S_ * S_ * 2;   // 268 MB
    (void)cur; (void)ws_size; (void)in_sizes; (void)n_in; (void)out_size;

    // 0. transpose/convert x, convert weights
    k_xt<<<dim3(S_ / 16, C_ / 16, B_), dim3(16, 16), 0, stream>>>(x, xt);
    k_cvt<<<(CQ_ * C_ + 255) / 256, 256, 0, stream>>>(wq, wqb, CQ_ * C_);
    k_cvt<<<(CQ_ * C_ + 255) / 256, 256, 0, stream>>>(wk, wkb, CQ_ * C_);
    k_cvt<<<(C_ * C_ + 255) / 256, 256, 0, stream>>>(wv, wvb, C_ * C_);

    // 1. Q and K projections: B*(S/16) = 2048 waves, 4 waves/block
    k_proj_qk<<<(B_ * (S_ / 16)) / 4, 128, 0, stream>>>(xt, wqb, bq, qt);
    k_proj_qk<<<(B_ * (S_ / 16)) / 4, 128, 0, stream>>>(xt, wkb, bk, kt);

    // 2. V projection: B*(C/16)*(S/64) = 16384 waves
    k_proj_v<<<(B_ * (C_ / 16) * (S_ / 64)) / 4, 128, 0, stream>>>(xt, wvb, bv, vbf);

    // 3. Energy: B*(S/16)*(S/64) = 131072 waves
    k_energy<<<(B_ * (S_ / 16) * (S_ / 64)) / 4, 128, 0, stream>>>(qt, kt, att);

    // 4. Softmax: one block per row; writes fp32 (output) + bf16 (for k_out)
    k_softmax<<<B_ * S_, 256, 0, stream>>>(att, attbf);

    // 5. Output GEMM + residual: B*(C/16)*(S/64) = 16384 waves
    k_out<<<(B_ * (C_ / 16) * (S_ / 64)) / 4, 128, 0, stream>>>(vbf, attbf, x, gamma, out);
}
